// Seq2Seq_20332375179562
// MI455X (gfx1250) — compile-verified, matrix operand order checked by reading, not run on hardware
//
#include <hip/hip_runtime.h>
#include <hip/hip_bf16.h>
#include <math.h>

#define VOCAB 32000
#define BATCH 64
#define FRAME_DIM 4096
#define HID 256
#define V_STEP 80
#define C_STEP 50
#define T1 (V_STEP + C_STEP - 1)   /* 129 */
#define DEC_T (C_STEP - 1)         /* 49 */

typedef __bf16 bf16_t;
typedef __attribute__((ext_vector_type(16))) __bf16 v16bf;
typedef __attribute__((ext_vector_type(8)))  __bf16 v8bf;
typedef __attribute__((ext_vector_type(8)))  float  v8f;

// ---- WMMA helpers (CDNA5 16x16x32 bf16, f32 accumulate) --------------------
// A-fragment (16x32, row-major source, stride ld): per ISA layout, lane L
// (r = L&15, kh = L>>4) holds K = kh*8+0..7 and K = 16+kh*8+0..7 of row r.
// B-fragment mirrors this with lane -> column; we feed rows of the [N,K]
// weight matrix, which makes the wmma compute X @ W^T directly.
__device__ __forceinline__ v16bf load_frag(const bf16_t* __restrict__ base, int ld) {
  const int lane = threadIdx.x & 31;
  const int r    = lane & 15;
  const int kh   = lane >> 4;
  const bf16_t* p = base + (size_t)r * ld + kh * 8;
  v8bf lo = *(const v8bf*)(p);
  v8bf hi = *(const v8bf*)(p + 16);
  return __builtin_shufflevector(lo, hi, 0,1,2,3,4,5,6,7,8,9,10,11,12,13,14,15);
}

__device__ __forceinline__ v8f wmma_bf16(v16bf a, v16bf b, v8f c) {
  return __builtin_amdgcn_wmma_f32_16x16x32_bf16(
      /*neg_a=*/false, a, /*neg_b=*/false, b,
      /*c_mod=*/(short)0, c, /*reuse_a=*/false, /*reuse_b=*/false);
}

// ---- utility kernels -------------------------------------------------------
__global__ void k_f32_to_bf16(const float* __restrict__ in, bf16_t* __restrict__ out, int n) {
  int i = blockIdx.x * blockDim.x + threadIdx.x;
  if (i < n) out[i] = (bf16_t)in[i];
}
__global__ void k_zero_bf16(bf16_t* __restrict__ p, int n) {
  int i = blockIdx.x * blockDim.x + threadIdx.x;
  if (i < n) p[i] = (bf16_t)0.0f;
}
__global__ void k_zero_f32(float* __restrict__ p, int n) {
  int i = blockIdx.x * blockDim.x + threadIdx.x;
  if (i < n) p[i] = 0.0f;
}
__global__ void k_set_i32(int* __restrict__ p, int n, int v) {
  int i = blockIdx.x * blockDim.x + threadIdx.x;
  if (i < n) p[i] = v;
}

// ---- frame projection: v = video @ w1^T + b1 -> bf16 vseq[B,T1,H] ----------
// C is [5120, 256]; one 16x16 tile per wave. K = 4096.
__global__ void k_proj(const bf16_t* __restrict__ vid, const bf16_t* __restrict__ w1,
                       const float* __restrict__ b1, bf16_t* __restrict__ vseq) {
  int wid   = (blockIdx.x * blockDim.x + threadIdx.x) >> 5;
  int tileN = wid & 15;      // 0..15
  int tileM = wid >> 4;      // 0..319
  const bf16_t* ab = vid + (size_t)tileM * 16 * FRAME_DIM;
  const bf16_t* bb = w1  + (size_t)tileN * 16 * FRAME_DIM;
  v8f acc = {};
  for (int k = 0; k < FRAME_DIM; k += 32)
    acc = wmma_bf16(load_frag(ab + k, FRAME_DIM), load_frag(bb + k, FRAME_DIM), acc);
  int lane = threadIdx.x & 31;
  int col  = tileN * 16 + (lane & 15);
  int kh   = lane >> 4;
  float bias = b1[col];
#pragma unroll
  for (int r = 0; r < 8; ++r) {
    int m = tileM * 16 + r + kh * 8;      // 0..5119
    int b = m / V_STEP, t = m % V_STEP;
    vseq[(size_t)b * T1 * HID + (size_t)t * HID + col] = (bf16_t)(acc[r] + bias);
  }
}

// ---- LSTM gate GEMM: gates[64,1024] = X[64,Kx]@Wih^T + H[64,256]@Whh^T -----
__global__ void k_gates(const bf16_t* __restrict__ x, int ldx, int Kx,
                        const bf16_t* __restrict__ wih,
                        const bf16_t* __restrict__ h, int ldh,
                        const bf16_t* __restrict__ whh,
                        float* __restrict__ gates) {
  int wid   = (blockIdx.x * blockDim.x + threadIdx.x) >> 5;
  int tileM = wid >> 6;     // 0..3
  int tileN = wid & 63;     // 0..63
  v8f acc = {};
  const bf16_t* ab = x + (size_t)tileM * 16 * ldx;
  const bf16_t* bb = wih + (size_t)tileN * 16 * Kx;
  for (int k = 0; k < Kx; k += 32)
    acc = wmma_bf16(load_frag(ab + k, ldx), load_frag(bb + k, Kx), acc);
  const bf16_t* ab2 = h + (size_t)tileM * 16 * ldh;
  const bf16_t* bb2 = whh + (size_t)tileN * 16 * HID;
#pragma unroll
  for (int k = 0; k < HID; k += 32)
    acc = wmma_bf16(load_frag(ab2 + k, ldh), load_frag(bb2 + k, HID), acc);
  int lane = threadIdx.x & 31;
  int col  = tileN * 16 + (lane & 15);
  int kh   = lane >> 4;
#pragma unroll
  for (int r = 0; r < 8; ++r) {
    int m = tileM * 16 + r + kh * 8;
    gates[(size_t)m * (4 * HID) + col] = acc[r];
  }
}

// ---- LSTM pointwise: c,h update from gates (i,f,g,o layout) ----------------
__global__ void k_lstm_pw(const float* __restrict__ gates,
                          const float* __restrict__ bih, const float* __restrict__ bhh,
                          float* __restrict__ c,
                          bf16_t* __restrict__ hb, int ld_hb) {
  int i = blockIdx.x * blockDim.x + threadIdx.x;   // BATCH*HID
  if (i >= BATCH * HID) return;
  int b = i >> 8, j = i & (HID - 1);
  const float* g = gates + (size_t)b * 4 * HID;
  float gi = g[j]           + bih[j]           + bhh[j];
  float gf = g[HID + j]     + bih[HID + j]     + bhh[HID + j];
  float gg = g[2*HID + j]   + bih[2*HID + j]   + bhh[2*HID + j];
  float go = g[3*HID + j]   + bih[3*HID + j]   + bhh[3*HID + j];
  float si = 1.0f / (1.0f + expf(-gi));
  float sf = 1.0f / (1.0f + expf(-gf));
  float so = 1.0f / (1.0f + expf(-go));
  float cn = sf * c[i] + si * tanhf(gg);
  float hn = so * tanhf(cn);
  c[i] = cn;
  hb[(size_t)b * ld_hb + j] = (bf16_t)hn;
}

// ---- x2 builders for LSTM2 -------------------------------------------------
__global__ void k_enc_x2(const bf16_t* __restrict__ h1t, int ldseq, bf16_t* __restrict__ x2) {
  int i = blockIdx.x * blockDim.x + threadIdx.x;   // BATCH*2H
  if (i >= BATCH * 2 * HID) return;
  int b = i >> 9, j = i & (2 * HID - 1);
  bf16_t v = (bf16_t)0.0f;
  if (j >= HID) v = h1t[(size_t)b * ldseq + (j - HID)];
  x2[i] = v;
}
__global__ void k_dec_x2(const bf16_t* __restrict__ emb, const int* __restrict__ tok,
                         const bf16_t* __restrict__ h1t, int ldseq, bf16_t* __restrict__ x2) {
  int i = blockIdx.x * blockDim.x + threadIdx.x;
  if (i >= BATCH * 2 * HID) return;
  int b = i >> 9, j = i & (2 * HID - 1);
  bf16_t v;
  if (j < HID) v = emb[(size_t)tok[b] * HID + j];
  else         v = h1t[(size_t)b * ldseq + (j - HID)];
  x2[i] = v;
}

// ---- logits: out[64,32000] = h2 @ w2^T + b2 --------------------------------
__global__ void k_logits(const bf16_t* __restrict__ h2b, const bf16_t* __restrict__ w2,
                         const float* __restrict__ b2, float* __restrict__ out, int ldo) {
  int wid   = (blockIdx.x * blockDim.x + threadIdx.x) >> 5;
  int tileM = wid & 3;       // 0..3
  int tileN = wid >> 2;      // 0..1999
  const bf16_t* ab = h2b + (size_t)tileM * 16 * HID;
  const bf16_t* bb = w2  + (size_t)tileN * 16 * HID;
  v8f acc = {};
#pragma unroll
  for (int k = 0; k < HID; k += 32) {
    __builtin_prefetch(bb + k + 128, 0, 1);   // global_prefetch_b8 on gfx1250
    acc = wmma_bf16(load_frag(ab + k, HID), load_frag(bb + k, HID), acc);
  }
  int lane = threadIdx.x & 31;
  int col  = tileN * 16 + (lane & 15);
  int kh   = lane >> 4;
  float bias = b2[col];
#pragma unroll
  for (int r = 0; r < 8; ++r) {
    int m = tileM * 16 + r + kh * 8;
    out[(size_t)m * ldo + col] = acc[r] + bias;
  }
}

// ---- argmax over vocab (first-max tie break, like jnp.argmax) --------------
__global__ void k_argmax(const float* __restrict__ logits, int ldl,
                         int* __restrict__ tok, float* __restrict__ cap, int t) {
  __shared__ float smax[256];
  __shared__ int   sidx[256];
  int b = blockIdx.x;
  const float* row = logits + (size_t)b * ldl;
  float best = -3.4e38f; int bi = 0;
  for (int v = threadIdx.x; v < VOCAB; v += blockDim.x) {
    float x = row[v];
    if (x > best) { best = x; bi = v; }
  }
  smax[threadIdx.x] = best; sidx[threadIdx.x] = bi;
  __syncthreads();
  for (int s = 128; s > 0; s >>= 1) {
    if (threadIdx.x < (unsigned)s) {
      float om = smax[threadIdx.x + s]; int oi = sidx[threadIdx.x + s];
      if (om > smax[threadIdx.x] || (om == smax[threadIdx.x] && oi < sidx[threadIdx.x])) {
        smax[threadIdx.x] = om; sidx[threadIdx.x] = oi;
      }
    }
    __syncthreads();
  }
  if (threadIdx.x == 0) {
    tok[b] = sidx[0];
    cap[(size_t)b * DEC_T + t] = (float)sidx[0];
  }
}

// ---------------------------------------------------------------------------
extern "C" void kernel_launch(void* const* d_in, const int* in_sizes, int n_in,
                              void* d_out, int out_size, void* d_ws, size_t ws_size,
                              hipStream_t stream) {
  const float* video  = (const float*)d_in[0];
  const float* w1     = (const float*)d_in[1];
  const float* b1     = (const float*)d_in[2];
  const float* w2     = (const float*)d_in[3];
  const float* b2     = (const float*)d_in[4];
  const float* l1_wih = (const float*)d_in[5];
  const float* l1_whh = (const float*)d_in[6];
  const float* l1_bih = (const float*)d_in[7];
  const float* l1_bhh = (const float*)d_in[8];
  const float* l2_wih = (const float*)d_in[9];
  const float* l2_whh = (const float*)d_in[10];
  const float* l2_bih = (const float*)d_in[11];
  const float* l2_bhh = (const float*)d_in[12];
  const float* emb    = (const float*)d_in[13];

  float* cap     = (float*)d_out;                      // [64,49] tokens as float
  float* capprob = (float*)d_out + BATCH * DEC_T;      // [64,49,32000]

  // workspace carving (256B aligned slices)
  char* w = (char*)d_ws;
  auto carve = [&](size_t bytes) -> char* {
    char* p = w; w += (bytes + 255) & ~(size_t)255; return p;
  };
  bf16_t* videoB  = (bf16_t*)carve((size_t)BATCH * V_STEP * FRAME_DIM * 2);
  bf16_t* w1b     = (bf16_t*)carve((size_t)HID * FRAME_DIM * 2);
  bf16_t* w2b     = (bf16_t*)carve((size_t)VOCAB * HID * 2);
  bf16_t* l1wihb  = (bf16_t*)carve((size_t)4 * HID * HID * 2);
  bf16_t* l1whhb  = (bf16_t*)carve((size_t)4 * HID * HID * 2);
  bf16_t* l2wihb  = (bf16_t*)carve((size_t)4 * HID * 2 * HID * 2);
  bf16_t* l2whhb  = (bf16_t*)carve((size_t)4 * HID * HID * 2);
  bf16_t* embb    = (bf16_t*)carve((size_t)VOCAB * HID * 2);
  bf16_t* vseq    = (bf16_t*)carve((size_t)BATCH * T1 * HID * 2);
  bf16_t* h1seq   = (bf16_t*)carve((size_t)BATCH * T1 * HID * 2);
  bf16_t* zbuf    = (bf16_t*)carve((size_t)BATCH * HID * 2);
  bf16_t* h2b     = (bf16_t*)carve((size_t)BATCH * HID * 2);
  bf16_t* x2      = (bf16_t*)carve((size_t)BATCH * 2 * HID * 2);
  float*  gatesB  = (float*)carve((size_t)BATCH * 4 * HID * 4);
  float*  c1      = (float*)carve((size_t)BATCH * HID * 4);
  float*  c2      = (float*)carve((size_t)BATCH * HID * 4);
  int*    tok     = (int*)carve((size_t)BATCH * 4);

  auto grid1d = [](int n) { return dim3((n + 255) / 256); };

  // ---- one-time conversions f32 -> bf16 ----
  { int n = BATCH * V_STEP * FRAME_DIM; k_f32_to_bf16<<<grid1d(n), 256, 0, stream>>>(video, videoB, n); }
  { int n = HID * FRAME_DIM;            k_f32_to_bf16<<<grid1d(n), 256, 0, stream>>>(w1, w1b, n); }
  { int n = VOCAB * HID;                k_f32_to_bf16<<<grid1d(n), 256, 0, stream>>>(w2, w2b, n); }
  { int n = 4 * HID * HID;              k_f32_to_bf16<<<grid1d(n), 256, 0, stream>>>(l1_wih, l1wihb, n); }
  { int n = 4 * HID * HID;              k_f32_to_bf16<<<grid1d(n), 256, 0, stream>>>(l1_whh, l1whhb, n); }
  { int n = 4 * HID * 2 * HID;          k_f32_to_bf16<<<grid1d(n), 256, 0, stream>>>(l2_wih, l2wihb, n); }
  { int n = 4 * HID * HID;              k_f32_to_bf16<<<grid1d(n), 256, 0, stream>>>(l2_whh, l2whhb, n); }
  { int n = VOCAB * HID;                k_f32_to_bf16<<<grid1d(n), 256, 0, stream>>>(emb, embb, n); }

  // ---- state init ----
  { int n = BATCH * T1 * HID; k_zero_bf16<<<grid1d(n), 256, 0, stream>>>(vseq, n); }  // zero pad t>=80
  { int n = BATCH * HID;      k_zero_bf16<<<grid1d(n), 256, 0, stream>>>(zbuf, n); }
  { int n = BATCH * HID;      k_zero_bf16<<<grid1d(n), 256, 0, stream>>>(h2b, n); }
  { int n = BATCH * HID;      k_zero_f32 <<<grid1d(n), 256, 0, stream>>>(c1, n); }
  { int n = BATCH * HID;      k_zero_f32 <<<grid1d(n), 256, 0, stream>>>(c2, n); }
  k_set_i32<<<1, 64, 0, stream>>>(tok, BATCH, 0 /*BOS*/);

  // ---- frame projection (WMMA GEMM, K=4096) ----
  k_proj<<<640, 256, 0, stream>>>(videoB, w1b, b1, vseq);

  const int LDSEQ = T1 * HID;

  // ---- LSTM1 over 129 steps ----
  for (int t = 0; t < T1; ++t) {
    const bf16_t* hprev = (t == 0) ? zbuf : (h1seq + (size_t)(t - 1) * HID);
    int ldh = (t == 0) ? HID : LDSEQ;
    k_gates<<<32, 256, 0, stream>>>(vseq + (size_t)t * HID, LDSEQ, HID,
                                    l1wihb, hprev, ldh, l1whhb, gatesB);
    k_lstm_pw<<<64, 256, 0, stream>>>(gatesB, l1_bih, l1_bhh, c1,
                                      h1seq + (size_t)t * HID, LDSEQ);
  }

  // ---- LSTM2 encoding phase: x2 = [zeros | h1seq[:,t]] for t in 0..79 ----
  for (int t = 0; t < V_STEP; ++t) {
    k_enc_x2<<<128, 256, 0, stream>>>(h1seq + (size_t)t * HID, LDSEQ, x2);
    k_gates<<<32, 256, 0, stream>>>(x2, 2 * HID, 2 * HID,
                                    l2wihb, h2b, HID, l2whhb, gatesB);
    k_lstm_pw<<<64, 256, 0, stream>>>(gatesB, l2_bih, l2_bhh, c2, h2b, HID);
  }

  // ---- autoregressive decode: 49 steps ----
  for (int t = 0; t < DEC_T; ++t) {
    k_dec_x2<<<128, 256, 0, stream>>>(embb, tok,
                                      h1seq + (size_t)(V_STEP + t) * HID, LDSEQ, x2);
    k_gates<<<32, 256, 0, stream>>>(x2, 2 * HID, 2 * HID,
                                    l2wihb, h2b, HID, l2whhb, gatesB);
    k_lstm_pw<<<64, 256, 0, stream>>>(gatesB, l2_bih, l2_bhh, c2, h2b, HID);
    k_logits<<<1000, 256, 0, stream>>>(h2b, w2b, b2,
                                       capprob + (size_t)t * VOCAB, DEC_T * VOCAB);
    k_argmax<<<64, 256, 0, stream>>>(capprob + (size_t)t * VOCAB, DEC_T * VOCAB,
                                     tok, cap, t);
  }
}
